// GaussianRasterizer_20315195310087
// MI455X (gfx1250) — matrix operand order repaired and checked by reading, not compiled, not run on hardware
//
#include <hip/hip_runtime.h>
#include <hip/hip_bf16.h>
#include <math.h>

typedef __attribute__((ext_vector_type(16))) _Float16 v16h;
typedef __attribute__((ext_vector_type(8)))  float    v8f;
typedef __attribute__((ext_vector_type(8)))  int      v8i;

#define NG   2048
#define IMW  128
#define IMH  128
#define FOCX 128.0f
#define FOCY 128.0f
#define LIMX 0.65f
#define LIMY 0.65f
#define FARZ 1e10f

__device__ __forceinline__ float clampf(float v, float lo, float hi) {
    return fminf(fmaxf(v, lo), hi);
}

// ---------------------------------------------------------------------------
// Kernel 1: per-Gaussian preprocessing (conic, color, depth, radii)
// ws layout (floats, field-major, stride NG):
//   0 px, 1 py, 2 con_a, 3 con_b, 4 con_c, 5 opacity,
//   6 rad_eff (valid ? radius : -1), 7 R, 8 G, 9 B, 10 invdepth, 11 depth
// ---------------------------------------------------------------------------
__global__ __launch_bounds__(256) void gs_preprocess(
    const float* __restrict__ means3D, const float* __restrict__ opac,
    const float* __restrict__ dc, const float* __restrict__ shs,
    const float* __restrict__ scales, const float* __restrict__ rots,
    const float* __restrict__ viewm, const float* __restrict__ projm,
    float* __restrict__ ws, int* __restrict__ radii_out)
{
    int i = blockIdx.x * blockDim.x + threadIdx.x;
    if (i >= NG) return;

    float mx = means3D[i*3+0], my = means3D[i*3+1], mz = means3D[i*3+2];
    float qr = rots[i*4+0], qx = rots[i*4+1], qy = rots[i*4+2], qz = rots[i*4+3];

    float R00 = 1.f-2.f*(qy*qy+qz*qz), R01 = 2.f*(qx*qy-qr*qz), R02 = 2.f*(qx*qz+qr*qy);
    float R10 = 2.f*(qx*qy+qr*qz), R11 = 1.f-2.f*(qx*qx+qz*qz), R12 = 2.f*(qy*qz-qr*qx);
    float R20 = 2.f*(qx*qz-qr*qy), R21 = 2.f*(qy*qz+qr*qx), R22 = 1.f-2.f*(qx*qx+qy*qy);

    float s0 = scales[i*3+0], s1 = scales[i*3+1], s2 = scales[i*3+2];
    float M00=R00*s0, M01=R01*s1, M02=R02*s2;
    float M10=R10*s0, M11=R11*s1, M12=R12*s2;
    float M20=R20*s0, M21=R21*s1, M22=R22*s2;
    float S00=M00*M00+M01*M01+M02*M02;
    float S01=M00*M10+M01*M11+M02*M12;
    float S02=M00*M20+M01*M21+M02*M22;
    float S11=M10*M10+M11*M11+M12*M12;
    float S12=M10*M20+M11*M21+M12*M22;
    float S22=M20*M20+M21*M21+M22*M22;

    // p_view = m @ V[:3,:3] + V[3,:3]  (row-vector convention, V row-major 4x4)
    float pv0 = mx*viewm[0] + my*viewm[4] + mz*viewm[8]  + viewm[12];
    float pv1 = mx*viewm[1] + my*viewm[5] + mz*viewm[9]  + viewm[13];
    float pv2 = mx*viewm[2] + my*viewm[6] + mz*viewm[10] + viewm[14];

    float ph0 = mx*projm[0] + my*projm[4] + mz*projm[8]  + projm[12];
    float ph1 = mx*projm[1] + my*projm[5] + mz*projm[9]  + projm[13];
    float ph3 = mx*projm[3] + my*projm[7] + mz*projm[11] + projm[15];
    float inv_w = 1.0f / (ph3 + 1e-7f);
    float ppx = ph0 * inv_w, ppy = ph1 * inv_w;

    float tz   = pv2;
    float invz = 1.0f / tz;
    float txc  = clampf(pv0*invz, -LIMX, LIMX) * tz;
    float tyc  = clampf(pv1*invz, -LIMY, LIMY) * tz;

    float J00 = FOCX*invz, J02 = -FOCX*txc*invz*invz;
    float J11 = FOCY*invz, J12 = -FOCY*tyc*invz*invz;

    // T_ = J @ V[:3,:3]^T : T_[r][k] = sum_j J[r][j] * V[k][j]
    float T00 = J00*viewm[0] + J02*viewm[2];
    float T01 = J00*viewm[4] + J02*viewm[6];
    float T02 = J00*viewm[8] + J02*viewm[10];
    float T10 = J11*viewm[1] + J12*viewm[2];
    float T11 = J11*viewm[5] + J12*viewm[6];
    float T12 = J11*viewm[9] + J12*viewm[10];

    float t00 = T00*S00 + T01*S01 + T02*S02;
    float t01 = T00*S01 + T01*S11 + T02*S12;
    float t02 = T00*S02 + T01*S12 + T02*S22;
    float t10 = T10*S00 + T11*S01 + T12*S02;
    float t11 = T10*S01 + T11*S11 + T12*S12;
    float t12 = T10*S02 + T11*S12 + T12*S22;
    float cov00 = t00*T00 + t01*T01 + t02*T02;
    float cov01 = t00*T10 + t01*T11 + t02*T12;
    float cov11 = t10*T10 + t11*T11 + t12*T12;

    float a = cov00 + 0.3f, b = cov01, c = cov11 + 0.3f;
    float det = a*c - b*b;
    bool  valid = (tz > 0.2f) && (det > 0.f);
    float det_inv = (det > 0.f) ? (1.0f / det) : 0.f;
    float con_a =  c * det_inv;
    float con_b = -b * det_inv;
    float con_c =  a * det_inv;
    float mid  = 0.5f * (a + c);
    float lam1 = mid + sqrtf(fmaxf(0.1f, mid*mid - det));
    float radius = ceilf(3.0f * sqrtf(lam1));

    radii_out[i] = valid ? (int)radius : 0;

    float px = ((ppx + 1.0f) * (float)IMW - 1.0f) * 0.5f;
    float py = ((ppy + 1.0f) * (float)IMH - 1.0f) * 0.5f;

    // SH evaluation (CAMPOS = 0)
    float nrm = sqrtf(mx*mx + my*my + mz*mz);
    float dxn = mx/nrm, dyn = my/nrm, dzn = mz/nrm;
    float xx = dxn*dxn, yy = dyn*dyn, zz = dzn*dzn;
    float xy = dxn*dyn, yz = dyn*dzn, xz = dxn*dzn;
    const float C0 = 0.28209479177387814f, C1 = 0.4886025119029199f;
    const float C20 = 1.0925484305920792f, C21 = -1.0925484305920792f,
                C22 = 0.31539156525252005f, C23 = -1.0925484305920792f,
                C24 = 0.5462742152960396f;
    const float C30 = -0.5900435899266435f, C31 = 2.890611442640554f,
                C32 = -0.4570457994644657f, C33 = 0.3731763325901154f,
                C34 = -0.4570457994644657f, C35 = 1.445305721320277f,
                C36 = -0.5900435899266435f;
    float colv[3];
    #pragma unroll
    for (int cc2 = 0; cc2 < 3; ++cc2) {
        const float* sh = shs + i*45 + cc2;
        float res = C0 * dc[i*3 + cc2];
        res += -C1*dyn*sh[0*3] + C1*dzn*sh[1*3] - C1*dxn*sh[2*3];
        res += C20*xy*sh[3*3] + C21*yz*sh[4*3] + C22*(2.f*zz-xx-yy)*sh[5*3]
             + C23*xz*sh[6*3] + C24*(xx-yy)*sh[7*3];
        res += C30*dyn*(3.f*xx-yy)*sh[8*3] + C31*xy*dzn*sh[9*3]
             + C32*dyn*(4.f*zz-xx-yy)*sh[10*3] + C33*dzn*(2.f*zz-3.f*xx-3.f*yy)*sh[11*3]
             + C34*dxn*(4.f*zz-xx-yy)*sh[12*3] + C35*dzn*(xx-yy)*sh[13*3]
             + C36*dxn*(xx-3.f*yy)*sh[14*3];
        colv[cc2] = fmaxf(res + 0.5f, 0.f);
    }

    float depth = valid ? tz : FARZ;
    ws[ 0*NG + i] = px;
    ws[ 1*NG + i] = py;
    ws[ 2*NG + i] = con_a;
    ws[ 3*NG + i] = con_b;
    ws[ 4*NG + i] = con_c;
    ws[ 5*NG + i] = opac[i];
    ws[ 6*NG + i] = valid ? radius : -1.0f;   // -1 kills |dx|<=rad when invalid
    ws[ 7*NG + i] = colv[0];
    ws[ 8*NG + i] = colv[1];
    ws[ 9*NG + i] = colv[2];
    ws[10*NG + i] = 1.0f / depth;
    ws[11*NG + i] = depth;
}

// ---------------------------------------------------------------------------
// Kernel 2: in-LDS bitonic sort by depth (index tie-break => stable-equivalent)
// then gather the SOA into sorted order at ws + 12*NG; ids at ws + 23*NG.
// ---------------------------------------------------------------------------
__global__ __launch_bounds__(1024) void gs_sort(float* __restrict__ ws)
{
    __shared__ float sk[NG];
    __shared__ int   sv[NG];
    int tid = threadIdx.x;
    const float* depth = ws + 11*NG;
    for (int e = tid; e < NG; e += 1024) { sk[e] = depth[e]; sv[e] = e; }
    __syncthreads();

    for (int k = 2; k <= NG; k <<= 1) {
        for (int j = k >> 1; j > 0; j >>= 1) {
            for (int e = tid; e < NG; e += 1024) {
                int ixj = e ^ j;
                if (ixj > e) {
                    float ka = sk[e], kb = sk[ixj];
                    int   va = sv[e], vb = sv[ixj];
                    bool up = ((e & k) == 0);
                    bool sw = up ? (ka > kb || (ka == kb && va > vb))
                                 : (ka < kb || (ka == kb && va < vb));
                    if (sw) { sk[e]=kb; sk[ixj]=ka; sv[e]=vb; sv[ixj]=va; }
                }
            }
            __syncthreads();
        }
    }

    float* sorted = ws + 12*NG;
    int*   sid    = (int*)(ws + 23*NG);
    for (int e = tid; e < NG; e += 1024) {
        int o = sv[e];
        #pragma unroll
        for (int f = 0; f < 11; ++f) sorted[f*NG + e] = ws[f*NG + o];
        sid[e] = o;
    }
}

// ---------------------------------------------------------------------------
// Kernel 3: tiled rasterizer. Block = 16x16 pixel tile, 8 waves.
// Wave w owns tile rows 2w (lanes 0-15) and 2w+1 (lanes 16-31): each lane
// computes the sequential f32 weights for exactly ONE pixel (no redundancy),
// packs them to f16, then a half-wave __shfl_xor(16) exchange assembles both
// WMMA A operands per the ISA layout. Color/invdepth accumulate on the matrix
// pipe via v_wmma_f32_16x16x32_f16, co-executing with the weight VALU stream.
// ---------------------------------------------------------------------------
__global__ __launch_bounds__(256) void gs_raster(const float* __restrict__ ws,
                                                 float* __restrict__ out)
{
    __shared__ float cdat[11*32];
    __shared__ int   cid[32];
    const float* sorted = ws + 12*NG;
    const int*   sid    = (const int*)(ws + 23*NG);

    int  tid  = threadIdx.x;
    int  wave = tid >> 5;
    int  lane = tid & 31;
    int  col  = lane & 15;
    bool hiw  = lane >= 16;

    int tx0  = (blockIdx.x & 7)  * 16;
    int ty0  = (blockIdx.x >> 3) * 16;
    int rowA = ty0 + wave*2;
    int rowB = rowA + 1;
    int myrow = hiw ? rowB : rowA;

    float X = (float)(tx0 + col);
    float Y = (float)myrow;

    float T = 1.f, maxw = 0.f;
    int   best = -1;
    v8f accA = {}; v8f accB = {};

    union HI { v16h h; v8i i; };

    for (int chnk = 0; chnk < NG/32; ++chnk) {
        __syncthreads();
        if (tid < 352) cdat[tid] = sorted[(tid >> 5)*NG + chnk*32 + (tid & 31)];
        if (tid < 32)  cid[tid]  = sid[chnk*32 + tid];
        __syncthreads();
        if (chnk + 1 < NG/32 && tid < 11)
            __builtin_prefetch(&sorted[tid*NG + (chnk + 1)*32], 0, 1);

        // ---- per-lane sequential weights for this lane's pixel -----------
        HI wlo, whi;   // wlo: K0..15, whi: K16..31 (f16, compile-time slots)
        #pragma unroll
        for (int k = 0; k < 32; ++k) {
            float gx  = cdat[0*32+k], gy = cdat[1*32+k];
            float ca  = cdat[2*32+k], cb = cdat[3*32+k], cc = cdat[4*32+k];
            float op  = cdat[5*32+k], rad = cdat[6*32+k];
            float dx  = X - gx;
            float dy  = Y - gy;
            float pw  = -0.5f*(ca*dx*dx + cc*dy*dy) - cb*dx*dy;
            float al  = fminf(0.99f, op * __expf(pw));
            bool  ins = (pw <= 0.f) && (al >= (1.0f/255.0f)) &&
                        (fabsf(dx) <= rad) && (fabsf(dy) <= rad);
            float ae  = ins ? al : 0.f;
            float w   = ae * T;
            if (w > maxw) { maxw = w; best = cid[k]; }
            T *= (1.f - ae);
            if (k < 16) wlo.h[k] = (_Float16)w;
            else        whi.h[k - 16] = (_Float16)w;
        }

        // ---- half-wave exchange: other = data of lane ^ 16 ---------------
        v8i olo, ohi;
        #pragma unroll
        for (int j = 0; j < 8; ++j) {
            olo[j] = __shfl_xor(wlo.i[j], 16, 32);
            ohi[j] = __shfl_xor(whi.i[j], 16, 32);
        }

        // ---- assemble WMMA A operands (ISA 16-bit A 16x32 layout) --------
        // row-A matrix: lanes 0-15 need own K0..7 & K16..23;
        //               lanes 16-31 need row-A pixel's K8..15 & K24..31 (from other)
        // row-B matrix: mirrored.
        HI aA, aB;
        #pragma unroll
        for (int j = 0; j < 4; ++j) {
            aA.i[j]     = hiw ? olo[4 + j]   : wlo.i[j];
            aA.i[4 + j] = hiw ? ohi[4 + j]   : whi.i[j];
            aB.i[j]     = hiw ? wlo.i[4 + j] : olo[j];
            aB.i[4 + j] = hiw ? whi.i[4 + j] : ohi[j];
        }

        // ---- B operand (32x16 f16): lane n<16 holds column n for K=0..7,16..23,
        // lane n+16 for K=8..15,24..31. Columns: 0=R 1=G 2=B 3=1/z, rest zero. ----
        v16h bm;
        int kb = hiw ? 8 : 0;
        #pragma unroll
        for (int s = 0; s < 16; ++s) {
            int kk = kb + s + (s >= 8 ? 8 : 0);
            float v = 0.f;
            if      (col == 0) v = cdat[ 7*32 + kk];
            else if (col == 1) v = cdat[ 8*32 + kk];
            else if (col == 2) v = cdat[ 9*32 + kk];
            else if (col == 3) v = cdat[10*32 + kk];
            bm[s] = (_Float16)v;
        }

        accA = __builtin_amdgcn_wmma_f32_16x16x32_f16(
                   false, aA.h, false, bm, (short)0, accA, false, false);
        accB = __builtin_amdgcn_wmma_f32_16x16x32_f16(
                   false, aB.h, false, bm, (short)0, accB, false, false);
    }

    // D layout: lane = channel column (N), VGPR r = pixel M (lane<16: M=r,
    // lane>=16: M=8+r). Channel lanes 0..3 & 16..19 write color/invdepth.
    float* colorOut = out;
    float* invdOut  = out + 3*IMH*IMW;
    int*   mgOut    = (int*)out + 4*IMH*IMW;

    if (col < 4) {
        #pragma unroll
        for (int r = 0; r < 8; ++r) {
            int m = (hiw ? 8 : 0) + r;
            int x = tx0 + m;
            if (col < 3) {
                colorOut[col*IMH*IMW + rowA*IMW + x] = accA[r];
                colorOut[col*IMH*IMW + rowB*IMW + x] = accB[r];
            } else {
                invdOut[rowA*IMW + x] = accA[r];
                invdOut[rowB*IMW + x] = accB[r];
            }
        }
    }
    // every lane owns exactly one pixel for argmax tracking
    mgOut[myrow*IMW + (tx0 + col)] = (maxw > 0.f) ? best : -1;
}

// ---------------------------------------------------------------------------
extern "C" void kernel_launch(void* const* d_in, const int* in_sizes, int n_in,
                              void* d_out, int out_size, void* d_ws, size_t ws_size,
                              hipStream_t stream)
{
    const float* means3D = (const float*)d_in[0];
    // d_in[1] = means2D (unused by the reference)
    const float* opac    = (const float*)d_in[2];
    const float* dc      = (const float*)d_in[3];
    const float* shs     = (const float*)d_in[4];
    const float* scales  = (const float*)d_in[5];
    const float* rots    = (const float*)d_in[6];
    const float* viewm   = (const float*)d_in[7];
    const float* projm   = (const float*)d_in[8];

    float* ws  = (float*)d_ws;
    float* out = (float*)d_out;
    // d_out element layout: color[3*128*128] | invdepth[128*128] |
    //                       mainGaussID[128*128] (int) | radii[2048] (int)
    int* radii_out = (int*)d_out + 4*IMH*IMW + IMH*IMW;

    gs_preprocess<<<NG/256, 256, 0, stream>>>(means3D, opac, dc, shs, scales,
                                              rots, viewm, projm, ws, radii_out);
    gs_sort<<<1, 1024, 0, stream>>>(ws);
    gs_raster<<<64, 256, 0, stream>>>(ws, out);
}
